// THLSTMCell_69758858821994
// MI455X (gfx1250) — matrix operand order repaired and checked by reading, not compiled, not run on hardware
//
#include <hip/hip_runtime.h>
#include <hip/hip_bf16.h>
#include <math.h>

// THLSTM cell, B=16384, I=H=512 on gfx1250 (MI455X).
// bf16 WMMA (v_wmma_f32_16x16x32_bf16), f32 accumulation.
// B tiles double-buffered in LDS via gfx1250 async global->LDS DMA (ASYNCcnt),
// raw split barriers (no waitcnt-0 fence), A fragments prefetched in registers.
// Gates kernel uses K-step = 64 (two 32-k sub-tiles per buffer) to halve
// barrier/async overhead and widen the schedulable region per step.

#define BSZ   16384
#define HDIM  512
#define KHX   1024          // K for GEMM1 (h_prev | x_t)
#define K2    1536          // K for GEMM2 (h_prev | x_t | s_t)
#define LDA   1536          // row stride of packed bf16 activation matrix
#define NGATE 2560          // 5 * H
#define PADK  40            // padded LDS row length (elements): 80B, conflict-free

typedef __bf16 bf16x16 __attribute__((ext_vector_type(16)));
typedef float  f32x8   __attribute__((ext_vector_type(8)));
typedef unsigned int u32x4 __attribute__((ext_vector_type(4)));

union Frag { bf16x16 bf; u32x4 u[2]; };

// ---- gfx1250 async global->LDS copy (ASYNCcnt) ----
__device__ __forceinline__ void async_load_b128(unsigned lds_off, const void* sbase,
                                                unsigned vbyte_off) {
    // GVS mode: mem = SGPR64 + VGPR32 ; VDST = per-lane LDS byte address
    asm volatile("global_load_async_to_lds_b128 %0, %1, %2"
                 :: "v"(lds_off), "v"(vbyte_off), "s"(sbase)
                 : "memory");
}

template <int N>
__device__ __forceinline__ void wait_asynccnt() {
    asm volatile("s_wait_asynccnt %0" :: "i"(N) : "memory");
}

// Raw workgroup split barrier: no implicit s_wait_loadcnt/dscnt 0 fence.
// Safe: LDS producers are async-DMA (waited via asynccnt before signaling);
// LDS consumers are ds_loads whose dscnt waits are forced by WMMA data deps.
__device__ __forceinline__ void block_barrier() {
    asm volatile("s_barrier_signal -1\n\ts_barrier_wait -1" ::: "memory");
}

// A fragment: 16x32 bf16 from row-major global memory, row stride lda (elements).
// lanes 0-15 -> row r, K {0..7, 16..23}; lanes 16-31 -> row r, K {8..15, 24..31}.
__device__ __forceinline__ bf16x16 load_a_frag(const __bf16* base, int lda, int lane) {
    const int half = lane >> 4;
    const int r    = lane & 15;
    const __bf16* rowp = base + (size_t)r * lda;
    Frag f;
    f.u[0] = *(const u32x4*)(rowp + half * 8);
    f.u[1] = *(const u32x4*)(rowp + 16 + half * 8);
    return f.bf;
}

// B fragment: 32x16 bf16 from LDS tile stored [n][k], row stride PADK elements.
// lanes 0-15 -> col n=r, K {0..7, 16..23}; lanes 16-31 -> col n=r, K {8..15, 24..31}.
__device__ __forceinline__ bf16x16 lds_b_frag(const __bf16* base, int lane) {
    const int half = lane >> 4;
    const int r    = lane & 15;
    const __bf16* p = base + r * PADK + half * 8;
    Frag f;
    f.u[0] = *(const u32x4*)(p);
    f.u[1] = *(const u32x4*)(p + 16);
    return f.bf;
}

__device__ __forceinline__ float sigmoidf_(float x) {
    return 1.0f / (1.0f + __expf(-x));
}

// 20-WMMA burst; B fragments pipelined 2 deep.
__device__ __forceinline__ void gates_burst(f32x8 (&acc)[5][4], const __bf16* Bbuf,
                                            bf16x16 a, int lane) {
    bf16x16 b0 = lds_b_frag(Bbuf + 0 * PADK, lane);
    bf16x16 b1 = lds_b_frag(Bbuf + 16 * PADK, lane);
#pragma unroll
    for (int t = 0; t < 20; ++t) {
        bf16x16 b2 = b0;
        if (t + 2 < 20) {
            const int n = ((t + 2) >> 2) * 64 + ((t + 2) & 3) * 16;
            b2 = lds_b_frag(Bbuf + n * PADK, lane);
        }
        acc[t >> 2][t & 3] = __builtin_amdgcn_wmma_f32_16x16x32_bf16(
            false, a, false, b0, (short)0, acc[t >> 2][t & 3], false, false);
        b0 = b1;
        b1 = b2;
    }
}

// 4-WMMA burst for the s-gate kernel.
__device__ __forceinline__ void s_burst(f32x8 (&acc)[4], const __bf16* Bbuf,
                                        bf16x16 a, int lane) {
    bf16x16 b0 = lds_b_frag(Bbuf + 0 * PADK, lane);
    bf16x16 b1 = lds_b_frag(Bbuf + 16 * PADK, lane);
#pragma unroll
    for (int t = 0; t < 4; ++t) {
        bf16x16 b2 = b0;
        if (t + 2 < 4) b2 = lds_b_frag(Bbuf + (t + 2) * 16 * PADK, lane);
        acc[t] = __builtin_amdgcn_wmma_f32_16x16x32_bf16(
            false, a, false, b0, (short)0, acc[t], false, false);
        b0 = b1;
        b1 = b2;
    }
}

// ---------------- prep kernels: fp32 -> bf16 packing / transposition ----------------

__global__ void __launch_bounds__(256) prep_hx(const float* __restrict__ h_prev,
                                               const float* __restrict__ x_t,
                                               __bf16* __restrict__ A2) {
    size_t idx = (size_t)blockIdx.x * 256 + threadIdx.x;
    if (idx >= (size_t)BSZ * HDIM) return;
    size_t row = idx >> 9;
    int    col = (int)(idx & 511);
    A2[row * LDA + col]       = (__bf16)h_prev[idx];
    A2[row * LDA + 512 + col] = (__bf16)x_t[idx];
}

__global__ void __launch_bounds__(256) prep_wsT(const float* __restrict__ Ws,
                                                __bf16* __restrict__ WsT) {
    int idx = blockIdx.x * 256 + threadIdx.x;
    if (idx >= HDIM * KHX) return;
    int n = idx >> 10;
    int k = idx & 1023;
    WsT[(size_t)n * KHX + k] = (__bf16)Ws[(size_t)k * HDIM + n];
}

__global__ void __launch_bounds__(256) prep_wgT(const float* __restrict__ Wg,
                                                __bf16* __restrict__ WgT) {
    int idx = blockIdx.x * 256 + threadIdx.x;
    if (idx >= NGATE * K2) return;
    int n = idx / K2;
    int k = idx - n * K2;
    WgT[(size_t)n * K2 + k] = (__bf16)Wg[(size_t)k * NGATE + n];
}

// ---------------- GEMM1: s_t = tanh([h|x] @ Ws[0:1024] + bs + delta*Ws[1024]) ----------------
// grid (B/128, 512/64), 256 thr = 8 waves. B tile 64x32 bf16 double-buffered in LDS.

__global__ void __launch_bounds__(256)
gemm_s_kernel(__bf16* __restrict__ A2, const __bf16* __restrict__ WsT,
              const float* __restrict__ Ws, const float* __restrict__ bs,
              const float* __restrict__ delta) {
    __shared__ __bf16 Bsh[2][64][PADK];
    const int tid  = threadIdx.x;
    const int wave = tid >> 5;
    const int lane = tid & 31;
    const int m0   = blockIdx.x * 128 + wave * 16;
    const int n0   = blockIdx.y * 64;

    const unsigned lds0       = (unsigned)(unsigned long long)(void*)&Bsh[0][0][0];
    const unsigned tile_bytes = 64 * PADK * 2;

    // cooperative B-tile load: 64 rows x 4 chunks(16B) = 256 chunks, 1 per thread
    const int cn    = tid >> 2;
    const int cpart = tid & 3;
    const unsigned lchunk = (unsigned)((cn * PADK + cpart * 8) * 2);
    const unsigned gbase  = (unsigned)((((n0 + cn) * KHX) + cpart * 8) * 2);

    const f32x8 vzero = {0.f, 0.f, 0.f, 0.f, 0.f, 0.f, 0.f, 0.f};
    f32x8 acc[4];
#pragma unroll
    for (int j = 0; j < 4; ++j) acc[j] = vzero;

    const __bf16* Abase = A2 + (size_t)m0 * LDA;
    const char*   Bsh0  = (const char*)&Bsh[0][0][0];

    // prologue: buffer 0 <- kk=0
    async_load_b128(lds0 + lchunk, WsT, gbase);
    bf16x16 a_cur = load_a_frag(Abase, LDA, lane);

    unsigned boff = 0;           // byte offset of the buffer being READ this step
    const int NS = KHX / 32;     // 32 steps
#pragma unroll 1
    for (int s = 0; s < NS - 1; ++s) {
        async_load_b128(lds0 + (boff ^ tile_bytes) + lchunk, WsT,
                        gbase + (unsigned)((s + 1) * 64));
        wait_asynccnt<1>();
        block_barrier();
        bf16x16 a_next = load_a_frag(Abase + (s + 1) * 32, LDA, lane);
        s_burst(acc, (const __bf16*)(Bsh0 + boff), a_cur, lane);
        block_barrier();
        a_cur = a_next;
        boff ^= tile_bytes;
    }
    wait_asynccnt<0>();
    block_barrier();
    s_burst(acc, (const __bf16*)(Bsh0 + boff), a_cur, lane);

    const int half = lane >> 4;
    const int r    = lane & 15;
#pragma unroll
    for (int j = 0; j < 4; ++j) {
        const int   col  = n0 + j * 16 + r;
        const float wrow = Ws[(size_t)KHX * HDIM + col];   // Ws row 1024 (delta coeff)
        const float bias = bs[col];
#pragma unroll
        for (int d = 0; d < 8; ++d) {
            const int row = m0 + half * 8 + d;
            float z = acc[j][d] + bias + delta[row] * wrow;
            A2[(size_t)row * LDA + 1024 + col] = (__bf16)tanhf(z);
        }
    }
}

// ---------------- GEMM2 + gates ----------------
// grid (B/128, 512/64); each wave: 5 gates x 4 n-subtiles = 20 accumulators.
// K-step = 64: each LDS buffer = two 32-k sub-tiles of 320x32 bf16 (all 5 gates),
// double-buffered (4 sub-tiles total, ~100KB) via async DMA.

__global__ void __launch_bounds__(256)
gemm_gates_kernel(const __bf16* __restrict__ A2, const __bf16* __restrict__ WgT,
                  const float* __restrict__ bg, const float* __restrict__ c_prev,
                  float* __restrict__ h_out, float* __restrict__ c_out) {
    __shared__ __bf16 Bsh[4][5 * 64][PADK];   // 4 sub-tiles: {buf0,buf1} x {klo,khi}
    const int tid  = threadIdx.x;
    const int wave = tid >> 5;
    const int lane = tid & 31;
    const int m0   = blockIdx.x * 128 + wave * 16;
    const int n0   = blockIdx.y * 64;

    const unsigned lds0       = (unsigned)(unsigned long long)(void*)&Bsh[0][0][0];
    const unsigned tile_bytes = 5 * 64 * PADK * 2;     // one 32-k sub-tile
    const unsigned buf_bytes  = 2 * tile_bytes;        // one K-step (64 k)

    const f32x8 vzero = {0.f, 0.f, 0.f, 0.f, 0.f, 0.f, 0.f, 0.f};
    f32x8 acc[5][4];
#pragma unroll
    for (int g = 0; g < 5; ++g)
#pragma unroll
        for (int j = 0; j < 4; ++j) acc[g][j] = vzero;

    const __bf16* Abase = A2 + (size_t)m0 * LDA;
    const char*   Bsh0  = (const char*)&Bsh[0][0][0];

    // cooperative sub-tile load: 320 rows x 4 chunks(16B) = 1280 chunks, 5 per thread.
    unsigned goff_c[5], loff_c[5];
#pragma unroll
    for (int c = 0; c < 5; ++c) {
        const int chunk = c * 256 + tid;
        const int n     = chunk >> 2;
        const int part  = chunk & 3;
        const int g     = n >> 6;
        const int col   = n & 63;
        goff_c[c] = (unsigned)((((g * HDIM + n0 + col) * K2) + part * 8) * 2);
        loff_c[c] = (unsigned)((n * PADK + part * 8) * 2);
    }

    // prologue: buffer 0 <- kk = 0..63 (both sub-tiles)
#pragma unroll
    for (int c = 0; c < 5; ++c)
        async_load_b128(lds0 + loff_c[c], WgT, goff_c[c]);
#pragma unroll
    for (int c = 0; c < 5; ++c)
        async_load_b128(lds0 + tile_bytes + loff_c[c], WgT, goff_c[c] + 64u);
    bf16x16 a_cur0 = load_a_frag(Abase, LDA, lane);
    bf16x16 a_cur1 = load_a_frag(Abase + 32, LDA, lane);

    unsigned boff = 0;           // byte offset of the buffer being READ this step
    const int NS = K2 / 64;      // 24 steps
#pragma unroll 1
    for (int s = 0; s < NS - 1; ++s) {
        const unsigned kbytes = (unsigned)((s + 1) * 128);
        const unsigned wr     = lds0 + (boff ^ buf_bytes);
#pragma unroll
        for (int c = 0; c < 5; ++c)
            async_load_b128(wr + loff_c[c], WgT, goff_c[c] + kbytes);
#pragma unroll
        for (int c = 0; c < 5; ++c)
            async_load_b128(wr + tile_bytes + loff_c[c], WgT, goff_c[c] + kbytes + 64u);
        wait_asynccnt<10>();
        block_barrier();
        bf16x16 a_next0 = load_a_frag(Abase + (s + 1) * 64, LDA, lane);
        bf16x16 a_next1 = load_a_frag(Abase + (s + 1) * 64 + 32, LDA, lane);
        gates_burst(acc, (const __bf16*)(Bsh0 + boff), a_cur0, lane);
        gates_burst(acc, (const __bf16*)(Bsh0 + boff + tile_bytes), a_cur1, lane);
        block_barrier();
        a_cur0 = a_next0;
        a_cur1 = a_next1;
        boff ^= buf_bytes;
    }
    wait_asynccnt<0>();
    block_barrier();
    gates_burst(acc, (const __bf16*)(Bsh0 + boff), a_cur0, lane);
    gates_burst(acc, (const __bf16*)(Bsh0 + boff + tile_bytes), a_cur1, lane);

    const int half = lane >> 4;
    const int r    = lane & 15;
#pragma unroll
    for (int j = 0; j < 4; ++j) {
        const int col = n0 + j * 16 + r;
        const float bf_ = bg[0 * HDIM + col];
        const float bi  = bg[1 * HDIM + col];
        const float bT  = bg[2 * HDIM + col];
        const float bu  = bg[3 * HDIM + col];
        const float bo  = bg[4 * HDIM + col];
#pragma unroll
        for (int d = 0; d < 8; ++d) {
            const int row = m0 + half * 8 + d;
            const float f_ = sigmoidf_(acc[0][j][d] + bf_);
            const float i_ = sigmoidf_(acc[1][j][d] + bi);
            const float T_ = sigmoidf_(acc[2][j][d] + bT);
            const float u_ = tanhf(acc[3][j][d] + bu);
            const float o_ = sigmoidf_(acc[4][j][d] + bo);
            const float s_ = (float)A2[(size_t)row * LDA + 1024 + col];
            const float cp = c_prev[(size_t)row * HDIM + col];
            const float c  = f_ * cp + i_ * u_ + T_ * s_;
            const float h  = o_ * tanhf(c);
            h_out[(size_t)row * HDIM + col] = h;
            c_out[(size_t)row * HDIM + col] = c;
        }
    }
}

// ---------------- launch ----------------

extern "C" void kernel_launch(void* const* d_in, const int* in_sizes, int n_in,
                              void* d_out, int out_size, void* d_ws, size_t ws_size,
                              hipStream_t stream) {
    (void)in_sizes; (void)n_in; (void)out_size; (void)ws_size;
    const float* x_t    = (const float*)d_in[0];
    const float* delta  = (const float*)d_in[1];
    const float* h_prev = (const float*)d_in[2];
    const float* c_prev = (const float*)d_in[3];
    const float* Ws     = (const float*)d_in[4];
    const float* bs     = (const float*)d_in[5];
    const float* Wg     = (const float*)d_in[6];
    const float* bg     = (const float*)d_in[7];

    float* h_out = (float*)d_out;
    float* c_out = h_out + (size_t)BSZ * HDIM;

    // workspace (bf16): A2 = B x 1536 packed [h|x|s] (48MB); WsT 512x1024 (1MB); WgT 2560x1536 (7.5MB)
    char*   ws  = (char*)d_ws;
    __bf16* A2  = (__bf16*)ws;
    __bf16* WsT = (__bf16*)(ws + (size_t)BSZ * LDA * 2);
    __bf16* WgT = (__bf16*)(ws + (size_t)BSZ * LDA * 2 + (size_t)HDIM * KHX * 2);

    {
        size_t n = (size_t)BSZ * HDIM;
        prep_hx<<<dim3((unsigned)((n + 255) / 256)), dim3(256), 0, stream>>>(h_prev, x_t, A2);
    }
    {
        int n = HDIM * KHX;
        prep_wsT<<<dim3((n + 255) / 256), dim3(256), 0, stream>>>(Ws, WsT);
    }
    {
        int n = NGATE * K2;
        prep_wgT<<<dim3((n + 255) / 256), dim3(256), 0, stream>>>(Wg, WgT);
    }

    gemm_s_kernel<<<dim3(BSZ / 128, HDIM / 64), dim3(256), 0, stream>>>(
        A2, WsT, Ws, bs, delta);

    gemm_gates_kernel<<<dim3(BSZ / 128, HDIM / 64), dim3(256), 0, stream>>>(
        A2, WgT, bg, c_prev, h_out, c_out);
}
